// RNNCellModel_64089501991291
// MI455X (gfx1250) — matrix operand order repaired and checked by reading, not compiled, not run on hardware
//
#include <hip/hip_runtime.h>
#include <math.h>
#include <stdint.h>

// ---------------------------------------------------------------------------
// Types for CDNA5 WMMA
// ---------------------------------------------------------------------------
typedef __attribute__((ext_vector_type(16))) __bf16 v16bf;
typedef __attribute__((ext_vector_type(8)))  float  v8f;

union FragBF {
    v16bf v;
    uint4 u[2];
};

// Model dims
#define BB   32
#define TT   200
#define XON  1024
#define XOFF 1024
#define NL4E 2048
#define NL4I 512
#define N23E 2048
#define N23I 512

#define NBLOCKS 16          // 16 WGs x 8 waves = 128 waves = 128 N-tiles

__device__ __forceinline__ unsigned short f2bf(float f) {
    // round-to-nearest-even fp32 -> bf16
    uint32_t x = __float_as_uint(f);
    uint32_t r = x + 0x7FFFu + ((x >> 16) & 1u);
    return (unsigned short)(r >> 16);
}

// ---------------------------------------------------------------------------
// One-time weight pack: dst is B-fragment ordered bf16 of sign*|[Wih | Whh]|^T
// Linear index: ((nt*nkt + kt)*32 + lane)*16 + slot
//   n = nt*16 + (lane&15),  k = kt*32 + ((lane&16)>>1) + (slot&7) + ((slot&8)<<1)
// ---------------------------------------------------------------------------
__global__ void pack_weights(const float* __restrict__ Wih,
                             const float* __restrict__ Whh,
                             unsigned short* __restrict__ dst,
                             int H, int Kih, int Kfull, float sign) {
    long long idx = (long long)blockIdx.x * blockDim.x + threadIdx.x;
    long long total = (long long)H * Kfull;
    if (idx >= total) return;
    int slot = (int)(idx & 15);
    int lane = (int)((idx >> 4) & 31);
    long long blk = idx >> 9;           // / 512
    int nkt = Kfull >> 5;
    int kt = (int)(blk % nkt);
    int nt = (int)(blk / nkt);
    int n = nt * 16 + (lane & 15);
    int k = kt * 32 + ((lane & 16) >> 1) + (slot & 7) + ((slot & 8) << 1);
    float w = (k < Kih) ? Wih[(long long)n * Kih + k]
                        : Whh[(long long)n * H + (k - Kih)];
    dst[idx] = f2bf(sign * fabsf(w));
}

__global__ void combine_bias(const float* __restrict__ b1,
                             const float* __restrict__ b2,
                             float* __restrict__ dst, int H) {
    int i = blockIdx.x * blockDim.x + threadIdx.x;
    if (i < H) dst[i] = b1[i] + b2[i];
}

__global__ void cvt_f32_bf16(const float* __restrict__ src,
                             unsigned short* __restrict__ dst, long long n) {
    long long i = (long long)blockIdx.x * blockDim.x + threadIdx.x;
    if (i < n) dst[i] = f2bf(src[i]);
}

__global__ void init_barrier(unsigned* bar) {
    if (threadIdx.x == 0) { bar[0] = 0u; bar[1] = 0u; }
}

// ---------------------------------------------------------------------------
// Device-wide sense-reversing barrier (all NBLOCKS blocks resident)
// ---------------------------------------------------------------------------
__device__ __forceinline__ void grid_sync(unsigned* bar) {
    __threadfence();
    __syncthreads();
    if (threadIdx.x == 0) {
        unsigned* cnt = bar;
        unsigned* gen = bar + 1;
        unsigned g = __hip_atomic_load(gen, __ATOMIC_RELAXED,
                                       __HIP_MEMORY_SCOPE_AGENT);
        unsigned a = __hip_atomic_fetch_add(cnt, 1u, __ATOMIC_ACQ_REL,
                                            __HIP_MEMORY_SCOPE_AGENT);
        if (a == (unsigned)(NBLOCKS - 1)) {
            __hip_atomic_store(cnt, 0u, __ATOMIC_RELAXED,
                               __HIP_MEMORY_SCOPE_AGENT);
            __hip_atomic_store(gen, g + 1u, __ATOMIC_RELEASE,
                               __HIP_MEMORY_SCOPE_AGENT);
        } else {
            while (__hip_atomic_load(gen, __ATOMIC_ACQUIRE,
                                     __HIP_MEMORY_SCOPE_AGENT) == g) {
                __builtin_amdgcn_s_sleep(2);
            }
        }
    }
    __syncthreads();
}

// ---------------------------------------------------------------------------
// One cell phase: D[32,H] = tanh(A[32,K] x W^T + bias)
// A is a 4-segment concat of bf16 row-major buffers (each len % 32 == 0).
// One N-tile per wave; both 16-row M-tiles per wave (B fragment reused).
// ---------------------------------------------------------------------------
struct Seg {
    const unsigned short* p;   // bf16 base
    int stride;                // elements between batch rows
    int len;                   // K extent of this segment
};

__device__ __forceinline__ v8f wmma_bf16(const FragBF& a, const FragBF& b,
                                         v8f c) {
    return __builtin_amdgcn_wmma_f32_16x16x32_bf16(
        false, a.v, false, b.v, (short)0, c, false, false);
}

__device__ __forceinline__ void cell_phase(
    const unsigned short* __restrict__ Bpack,
    const float* __restrict__ bias,
    float* __restrict__ out,                 // [B,T,H] slice base
    unsigned short* __restrict__ state_bf,   // next-parity bf16 state [B,H]
    int H, int Kfull, int t, const Seg segs[4], int gw, int lane) {
    const int ntile = gw;
    if (ntile * 16 >= H) return;             // wave-uniform: EXEC stays full

    const int mrow  = lane & 15;
    const int khalf = (lane & 16) >> 1;      // 0 / 8
    const int nkt   = Kfull >> 5;
    const unsigned short* bp =
        Bpack + ((long long)ntile * nkt * 32 + lane) * 16;

    v8f acc0 = {};
    v8f acc1 = {};
    long long bofs = 0;
    #pragma unroll
    for (int s = 0; s < 4; ++s) {
        const unsigned short* a0p =
            segs[s].p + (long long)mrow * segs[s].stride;
        const unsigned short* a1p =
            segs[s].p + (long long)(mrow + 16) * segs[s].stride;
        const int nk = segs[s].len >> 5;
        for (int kt = 0; kt < nk; ++kt) {
            const int k0 = kt * 32 + khalf;
            FragBF a0, a1, b;
            a0.u[0] = *reinterpret_cast<const uint4*>(a0p + k0);
            a0.u[1] = *reinterpret_cast<const uint4*>(a0p + k0 + 16);
            a1.u[0] = *reinterpret_cast<const uint4*>(a1p + k0);
            a1.u[1] = *reinterpret_cast<const uint4*>(a1p + k0 + 16);
            const unsigned short* bkt = bp + bofs;
            b.u[0] = *reinterpret_cast<const uint4*>(bkt);
            b.u[1] = *reinterpret_cast<const uint4*>(bkt + 8);
            __builtin_prefetch(bkt + 8 * 512, 0, 1);
            acc0 = wmma_bf16(a0, b, acc0);
            acc1 = wmma_bf16(a1, b, acc1);
            bofs += 512;
        }
    }

    const int n  = ntile * 16 + (lane & 15);
    const float bn = bias[n];
    const int mbase = (lane & 16) >> 1;      // rows 0-7 or 8-15 of each M tile
    #pragma unroll
    for (int r = 0; r < 8; ++r) {
        const int m0 = mbase + r;
        const float v0 = tanhf(acc0[r] + bn);
        out[((long long)m0 * TT + t) * H + n] = v0;
        state_bf[(long long)m0 * H + n] = f2bf(v0);
        const int m1 = m0 + 16;
        const float v1 = tanhf(acc1[r] + bn);
        out[((long long)m1 * TT + t) * H + n] = v1;
        state_bf[(long long)m1 * H + n] = f2bf(v1);
    }
}

// ---------------------------------------------------------------------------
// Persistent RNN kernel: the whole T=200 recurrence in one launch.
// ---------------------------------------------------------------------------
struct Params {
    const unsigned short *wp4e, *wp4i, *wp23e, *wp23i;
    const float *b4e, *b4i, *b23e, *b23i;
    float *o4e, *o4i, *o23e, *o23i;
    const unsigned short *xon, *xoff;        // bf16 [B,T,1024]
    unsigned short *s4e[2], *s4i[2], *s23e[2], *s23i[2];  // parity buffers
    unsigned* bar;
};

__global__ __launch_bounds__(256) void rnn_persistent(Params P) {
    const int lane = threadIdx.x & 31;
    const int gw   = blockIdx.x * 8 + (threadIdx.x >> 5);   // global wave id

    int p = 0;  // parity: read h_prev from [p], write h_new to [1-p]
    for (int t = 0; t < TT; ++t) {
        const unsigned short* xon_t  = P.xon  + (long long)t * XON;
        const unsigned short* xoff_t = P.xoff + (long long)t * XOFF;
        const int q = 1 - p;

        {   // cell 4e: [x_on | x_off | h4i(old) | h4e(old)] -> h4e(new)
            Seg segs[4] = {{xon_t, TT * XON, XON}, {xoff_t, TT * XOFF, XOFF},
                           {P.s4i[p], NL4I, NL4I}, {P.s4e[p], NL4E, NL4E}};
            cell_phase(P.wp4e, P.b4e, P.o4e, P.s4e[q],
                       NL4E, XON + XOFF + NL4I + NL4E, t, segs, gw, lane);
        }
        grid_sync(P.bar);

        {   // cell 4i: [x_on | x_off | h4e(new) | h4i(old)] -> h4i(new)
            Seg segs[4] = {{xon_t, TT * XON, XON}, {xoff_t, TT * XOFF, XOFF},
                           {P.s4e[q], NL4E, NL4E}, {P.s4i[p], NL4I, NL4I}};
            cell_phase(P.wp4i, P.b4i, P.o4i, P.s4i[q],
                       NL4I, XON + XOFF + NL4E + NL4I, t, segs, gw, lane);
        }
        grid_sync(P.bar);

        {   // cell 23e: [h4i(new) | h4e(new) | h23i(old) | h23e(old)]
            Seg segs[4] = {{P.s4i[q], NL4I, NL4I}, {P.s4e[q], NL4E, NL4E},
                           {P.s23i[p], N23I, N23I}, {P.s23e[p], N23E, N23E}};
            cell_phase(P.wp23e, P.b23e, P.o23e, P.s23e[q],
                       N23E, NL4I + NL4E + N23I + N23E, t, segs, gw, lane);
        }
        grid_sync(P.bar);

        {   // cell 23i: [h4i(new) | h4e(new) | h23e(new) | h23i(old)]
            Seg segs[4] = {{P.s4i[q], NL4I, NL4I}, {P.s4e[q], NL4E, NL4E},
                           {P.s23e[q], N23E, N23E}, {P.s23i[p], N23I, N23I}};
            cell_phase(P.wp23i, P.b23i, P.o23i, P.s23i[q],
                       N23I, NL4I + NL4E + N23E + N23I, t, segs, gw, lane);
        }
        grid_sync(P.bar);

        p = q;
    }
}

// ---------------------------------------------------------------------------
// Host-side orchestration  (16 launches total, all on `stream`)
// ---------------------------------------------------------------------------
extern "C" void kernel_launch(void* const* d_in, const int* in_sizes, int n_in,
                              void* d_out, int out_size, void* d_ws, size_t ws_size,
                              hipStream_t stream) {
    (void)in_sizes; (void)n_in; (void)out_size; (void)ws_size;

    const float* x_on  = (const float*)d_in[0];
    const float* x_off = (const float*)d_in[1];
    const float* h4e0  = (const float*)d_in[2];
    const float* h4i0  = (const float*)d_in[3];
    const float* h23e0 = (const float*)d_in[4];
    const float* h23i0 = (const float*)d_in[5];
    const float* Wih4e = (const float*)d_in[6];
    const float* Whh4e = (const float*)d_in[7];
    const float* bih4e = (const float*)d_in[8];
    const float* bhh4e = (const float*)d_in[9];
    const float* Wih4i = (const float*)d_in[10];
    const float* Whh4i = (const float*)d_in[11];
    const float* bih4i = (const float*)d_in[12];
    const float* bhh4i = (const float*)d_in[13];
    const float* Wih23e = (const float*)d_in[14];
    const float* Whh23e = (const float*)d_in[15];
    const float* bih23e = (const float*)d_in[16];
    const float* bhh23e = (const float*)d_in[17];
    const float* Wih23i = (const float*)d_in[18];
    const float* Whh23i = (const float*)d_in[19];
    const float* bih23i = (const float*)d_in[20];
    const float* bhh23i = (const float*)d_in[21];

    const int K4e  = XON + XOFF + NL4I + NL4E;     // 4608
    const int K4i  = XON + XOFF + NL4E + NL4I;     // 4608
    const int K23e = NL4I + NL4E + N23I + N23E;    // 5120
    const int K23i = NL4I + NL4E + N23E + N23I;    // 5120
    const int Kih4e = 2048 + 512, Kih4i = 2048 + 2048;
    const int Kih23e = 2560 + 512, Kih23i = 2560 + 2048;

    // Workspace partition
    size_t off = 0;
    auto take = [&](size_t bytes) -> char* {
        char* ptr = (char*)d_ws + off;
        off += (bytes + 255) & ~(size_t)255;
        return ptr;
    };
    unsigned short* wp4e  = (unsigned short*)take((size_t)K4e  * NL4E * 2);
    unsigned short* wp4i  = (unsigned short*)take((size_t)K4i  * NL4I * 2);
    unsigned short* wp23e = (unsigned short*)take((size_t)K23e * N23E * 2);
    unsigned short* wp23i = (unsigned short*)take((size_t)K23i * N23I * 2);
    float* bias4e  = (float*)take(NL4E * 4);
    float* bias4i  = (float*)take(NL4I * 4);
    float* bias23e = (float*)take(N23E * 4);
    float* bias23i = (float*)take(N23I * 4);
    unsigned short* xon_bf  = (unsigned short*)take((size_t)BB * TT * XON * 2);
    unsigned short* xoff_bf = (unsigned short*)take((size_t)BB * TT * XOFF * 2);
    unsigned short* s4e[2], *s4i[2], *s23e[2], *s23i[2];
    for (int i = 0; i < 2; ++i) {
        s4e[i]  = (unsigned short*)take((size_t)BB * NL4E * 2);
        s4i[i]  = (unsigned short*)take((size_t)BB * NL4I * 2);
        s23e[i] = (unsigned short*)take((size_t)BB * N23E * 2);
        s23i[i] = (unsigned short*)take((size_t)BB * N23I * 2);
    }
    unsigned* bar = (unsigned*)take(256);

    // Output slices (o4e, o4i, o23e, o23i) flattened in return order
    float* o4e  = (float*)d_out;
    float* o4i  = o4e  + (long long)BB * TT * NL4E;
    float* o23e = o4i  + (long long)BB * TT * NL4I;
    float* o23i = o23e + (long long)BB * TT * N23E;

    // ---- one-time preprocessing ----
    auto launch_pack_w = [&](const float* Wih, const float* Whh,
                             unsigned short* dst, int H, int Kih, int Kfull,
                             float sign) {
        long long total = (long long)H * Kfull;
        pack_weights<<<(int)((total + 255) / 256), 256, 0, stream>>>(
            Wih, Whh, dst, H, Kih, Kfull, sign);
    };
    launch_pack_w(Wih4e,  Whh4e,  wp4e,  NL4E, Kih4e,  K4e,   1.0f);
    launch_pack_w(Wih4i,  Whh4i,  wp4i,  NL4I, Kih4i,  K4i,  -1.0f);
    launch_pack_w(Wih23e, Whh23e, wp23e, N23E, Kih23e, K23e,  1.0f);
    launch_pack_w(Wih23i, Whh23i, wp23i, N23I, Kih23i, K23i, -1.0f);

    combine_bias<<<(NL4E + 255) / 256, 256, 0, stream>>>(bih4e,  bhh4e,  bias4e,  NL4E);
    combine_bias<<<(NL4I + 255) / 256, 256, 0, stream>>>(bih4i,  bhh4i,  bias4i,  NL4I);
    combine_bias<<<(N23E + 255) / 256, 256, 0, stream>>>(bih23e, bhh23e, bias23e, N23E);
    combine_bias<<<(N23I + 255) / 256, 256, 0, stream>>>(bih23i, bhh23i, bias23i, N23I);

    auto cvt = [&](const float* src, unsigned short* dst, long long n) {
        cvt_f32_bf16<<<(int)((n + 255) / 256), 256, 0, stream>>>(src, dst, n);
    };
    cvt(x_on,  xon_bf,  (long long)BB * TT * XON);
    cvt(x_off, xoff_bf, (long long)BB * TT * XOFF);
    cvt(h4e0,  s4e[0],  (long long)BB * NL4E);
    cvt(h4i0,  s4i[0],  (long long)BB * NL4I);
    cvt(h23e0, s23e[0], (long long)BB * N23E);
    cvt(h23i0, s23i[0], (long long)BB * N23I);

    init_barrier<<<1, 32, 0, stream>>>(bar);

    // ---- single persistent kernel runs the whole recurrence ----
    Params P;
    P.wp4e = wp4e;  P.wp4i = wp4i;  P.wp23e = wp23e;  P.wp23i = wp23i;
    P.b4e = bias4e; P.b4i = bias4i; P.b23e = bias23e; P.b23i = bias23i;
    P.o4e = o4e;    P.o4i = o4i;    P.o23e = o23e;    P.o23i = o23i;
    P.xon = xon_bf; P.xoff = xoff_bf;
    for (int i = 0; i < 2; ++i) {
        P.s4e[i] = s4e[i];   P.s4i[i] = s4i[i];
        P.s23e[i] = s23e[i]; P.s23i[i] = s23i[i];
    }
    P.bar = bar;

    rnn_persistent<<<NBLOCKS, 256, 0, stream>>>(P);
}
